// DecoderLayer_70497593196787
// MI455X (gfx1250) — compile-verified, matrix operand order checked by reading, not run on hardware
//
#include <hip/hip_runtime.h>
#include <hip/hip_bf16.h>

// ---------------------------------------------------------------------------
// Decoder layer for MI455X (gfx1250).
// All GEMMs: v_wmma_f32_16x16x32_bf16, operands staged to LDS by the Tensor
// Data Mover (tensor_load_to_lds + s_wait_tensorcnt), double-buffered.
// B=2, S=2048, D=2048, H=16, DH=128, F=8192.
// d_out = [ out (B*S*D f32) | attn (B*H*S*S f32) ]
// ---------------------------------------------------------------------------

typedef __attribute__((ext_vector_type(16))) __bf16        v16bf;
typedef __attribute__((ext_vector_type(8)))  __bf16        v8bf;
typedef __attribute__((ext_vector_type(8)))  float         v8f;
typedef __attribute__((ext_vector_type(4)))  unsigned int  v4u;
typedef __attribute__((ext_vector_type(8)))  int           v8i;
typedef __attribute__((ext_vector_type(4)))  int           v4i;

#define TB 2
#define TS 2048
#define TD 2048
#define THH 16
#define TDH 128
#define TF 8192

#define BK   64          // K elements staged per tile
#define LDSK 72          // padded LDS row stride (144 B, 16B aligned)

// ---------------------------------------------------------------------------
// Issue one TDM 2D tile load: tile_dim0=64 elems (K, contiguous, bf16),
// tile_dim1=128 rows, row stride = strideElems, LDS pad: every 32 dwords
// (one 128B row) insert 4 dwords -> LDS row stride 72 bf16 elements.
// D# layout per CDNA5 ISA 8.3/8.4. This toolchain: 6-arg builtin
// (g0, g1, g2, g3, extra_g, cpol).
// ---------------------------------------------------------------------------
__device__ __forceinline__ void tdm_tile_load(unsigned ldsOff, const void* gp,
                                              unsigned long long strideElems)
{
    const unsigned long long ga = (unsigned long long)(size_t)gp;
    v4u g0;
    g0[0] = 1u;                                           // count=1
    g0[1] = ldsOff;                                       // lds_addr
    g0[2] = (unsigned)ga;                                 // global_addr[31:0]
    g0[3] = (unsigned)((ga >> 32) & 0x01FFFFFFull)        // global_addr[56:32]
          | 0x80000000u;                                  // type=2 (image)

    const unsigned td0 = 1u << 20, td1 = 1u << 20;        // huge dims: no OOB clip
    v8i g1;
    g1[0] = (int)((1u << 16)      // data_size = 2 bytes
                | (1u << 20)      // pad_enable
                | (4u << 22)      // pad_interval: 32 dwords (=64 bf16 = one row)
                | (3u << 25));    // pad_amount: 4 dwords (=8 bf16)
    g1[1] = (int)((td0 & 0xFFFFu) << 16);                 // tensor_dim0[15:0]
    g1[2] = (int)(((td0 >> 16) & 0xFFFFu) | ((td1 & 0xFFFFu) << 16));
    g1[3] = (int)(((td1 >> 16) & 0xFFFFu) | ((unsigned)BK << 16)); // tile_dim0=64
    g1[4] = (int)128u;                                    // tile_dim1=128, tile_dim2=0
    g1[5] = (int)(unsigned)(strideElems & 0xFFFFFFFFull); // tensor_dim0_stride lo
    g1[6] = (int)(unsigned)((strideElems >> 32) & 0xFFFFull);
    g1[7] = 0;

    const v4i z4 = {0, 0, 0, 0};
    const v8i z8 = {0, 0, 0, 0, 0, 0, 0, 0};
    __builtin_amdgcn_tensor_load_to_lds(g0, g1, z4, z4, z8, 0);
}

// ---------------------------------------------------------------------------
// Tiled WMMA GEMM, 128x128 block tile, K-step 64, 256 threads = 8 waves (2x4),
// each wave: 4x2 16x16 f32 accumulators. A: [m][k]; B: [n][k] (both K-major).
//   ABF16:   A is bf16 in global, staged by TDM. Else f32, staged sync w/ cvt.
//   CAUSAL:  skip 128x128 tiles above the diagonal (scores).
//   CAUSALK: limit K loop to m0+128 (attn rows zero beyond diag; attn*V).
//   EPI:     0 = f32 store, 1 = bf16 store, 2 = bf16 transposed store.
// ---------------------------------------------------------------------------
template <bool ABF16, bool RELU, bool CAUSAL, bool CAUSALK, int EPI>
__global__ __launch_bounds__(256)
void gemm_wmma(const void* __restrict__ Agv, long long lda, long long sAb, long long sAh,
               const __bf16* __restrict__ Bg, long long ldb, long long sBb, long long sBh,
               void* __restrict__ Cgv,       long long ldc, long long sCb, long long sCh,
               int Kdim, int HH, float scale)
{
    const int bx = blockIdx.x, by = blockIdx.y, bz = blockIdx.z;
    if (CAUSAL && bx > by) return;

    const int zb = bz / HH, zh = bz % HH;
    const size_t zA = (size_t)zb * sAb + (size_t)zh * sAh;
    const size_t zC = (size_t)zb * sCb + (size_t)zh * sCh;
    Bg += (size_t)zb * sBb + (size_t)zh * sBh;

    const __bf16* A16 = ABF16 ? ((const __bf16*)Agv + zA) : nullptr;
    const float*  A32 = ABF16 ? nullptr : ((const float*)Agv + zA);

    const int m0 = by * 128;
    const int n0 = bx * 128;

    __shared__ __align__(16) __bf16 Als[2][128][LDSK];
    __shared__ __align__(16) __bf16 Bls[2][128][LDSK];

    const int tid  = threadIdx.x;
    const int lane = tid & 31;
    const int wave = tid >> 5;
    const int wm   = wave >> 2;          // 0..1 -> rows wm*64..+63
    const int wn   = wave & 3;           // 0..3 -> cols wn*32..+31
    const int l15  = lane & 15;
    const int kb   = (lane >> 4) << 3;   // 0 / 8 per ISA 16-bit operand layout

    v8f acc[4][2] = {};

    const int kend = CAUSALK ? ((m0 + 128 < Kdim) ? (m0 + 128) : Kdim) : Kdim;

    auto stageTile = [&](int buf, int k0) {
        if (tid == 0)                                          // TDM: B tile always
            tdm_tile_load((unsigned)(size_t)&Bls[buf][0][0],
                          Bg + (size_t)n0 * ldb + k0, (unsigned long long)ldb);
        if constexpr (ABF16) {
            if (tid == 0)                                      // TDM: A tile
                tdm_tile_load((unsigned)(size_t)&Als[buf][0][0],
                              A16 + (size_t)m0 * lda + k0, (unsigned long long)lda);
        } else {
            // synchronous f32 -> bf16 staging, all loads issued before any use
            float4 t[8];
#pragma unroll
            for (int i = 0; i < 8; ++i) {
                const int s = tid + (i << 8);                  // 2048 float4 slots
                const int m = s >> 4, kv = (s & 15) << 2;
                t[i] = *reinterpret_cast<const float4*>(
                    A32 + (size_t)(m0 + m) * lda + k0 + kv);
            }
#pragma unroll
            for (int i = 0; i < 8; ++i) {
                const int s = tid + (i << 8);
                const int m = s >> 4, kv = (s & 15) << 2;
                Als[buf][m][kv + 0] = (__bf16)t[i].x;
                Als[buf][m][kv + 1] = (__bf16)t[i].y;
                Als[buf][m][kv + 2] = (__bf16)t[i].z;
                Als[buf][m][kv + 3] = (__bf16)t[i].w;
            }
        }
    };

    stageTile(0, 0);                                           // prologue

    int buf = 0;
    for (int k0 = 0; k0 < kend; k0 += BK, buf ^= 1) {
        if (wave == 0) __builtin_amdgcn_s_wait_tensorcnt(0);   // DMA done (wave0)
        __syncthreads();                                       // LDS visible to all

        if (k0 + BK < kend) stageTile(buf ^ 1, k0 + BK);       // overlap next DMA

#pragma unroll
        for (int ks = 0; ks < 2; ++ks) {                       // two 32-wide substeps
            const int kc = (ks << 5) + kb;
            v16bf af[4];
#pragma unroll
            for (int m = 0; m < 4; ++m) {
                const int r = wm * 64 + m * 16 + l15;
                const v8bf lo = *reinterpret_cast<const v8bf*>(&Als[buf][r][kc]);
                const v8bf hi = *reinterpret_cast<const v8bf*>(&Als[buf][r][kc + 16]);
                af[m] = __builtin_shufflevector(lo, hi, 0, 1, 2, 3, 4, 5, 6, 7,
                                                        8, 9, 10, 11, 12, 13, 14, 15);
            }
            v16bf bfr[2];
#pragma unroll
            for (int n = 0; n < 2; ++n) {
                const int c = wn * 32 + n * 16 + l15;
                const v8bf lo = *reinterpret_cast<const v8bf*>(&Bls[buf][c][kc]);
                const v8bf hi = *reinterpret_cast<const v8bf*>(&Bls[buf][c][kc + 16]);
                bfr[n] = __builtin_shufflevector(lo, hi, 0, 1, 2, 3, 4, 5, 6, 7,
                                                         8, 9, 10, 11, 12, 13, 14, 15);
            }
#pragma unroll
            for (int m = 0; m < 4; ++m)
#pragma unroll
                for (int n = 0; n < 2; ++n)
                    acc[m][n] = __builtin_amdgcn_wmma_f32_16x16x32_bf16(
                        false, af[m], false, bfr[n], (short)0, acc[m][n], false, false);
        }
    }

    // epilogue: C/D layout -> lanes 0-15 M=r, lanes 16-31 M=r+8
    const int roff = (lane >> 4) << 3;
#pragma unroll
    for (int m = 0; m < 4; ++m) {
#pragma unroll
        for (int n = 0; n < 2; ++n) {
            const int col = n0 + wn * 32 + n * 16 + l15;
#pragma unroll
            for (int r = 0; r < 8; ++r) {
                const int row = m0 + wm * 64 + m * 16 + r + roff;
                float v = acc[m][n][r] * scale;
                if (RELU) v = fmaxf(v, 0.0f);
                if constexpr (EPI == 0)
                    ((float*)Cgv + zC)[(size_t)row * ldc + col] = v;
                else if constexpr (EPI == 1)
                    ((__bf16*)Cgv + zC)[(size_t)row * ldc + col] = (__bf16)v;
                else
                    ((__bf16*)Cgv + zC)[(size_t)col * ldc + row] = (__bf16)v;
            }
        }
    }
}

// ---------------------------------------------------------------------------
// f32 -> bf16 elementwise (vectorized), one-time prep for x
// ---------------------------------------------------------------------------
__global__ __launch_bounds__(256)
void cvt_bf16_kernel(const float* __restrict__ in, __bf16* __restrict__ out,
                     long long n4)
{
    const long long i = (long long)blockIdx.x * 256 + threadIdx.x;
    if (i < n4) {
        const float4 f = reinterpret_cast<const float4*>(in)[i];
        out[i * 4 + 0] = (__bf16)f.x;
        out[i * 4 + 1] = (__bf16)f.y;
        out[i * 4 + 2] = (__bf16)f.z;
        out[i * 4 + 3] = (__bf16)f.w;
    }
}

// ---------------------------------------------------------------------------
// f32 [R][C] -> bf16 transposed [C][R] (weights become K-major for TDM)
// ---------------------------------------------------------------------------
__global__ __launch_bounds__(256)
void transpose_cvt_kernel(const float* __restrict__ in, __bf16* __restrict__ out,
                          int R, int C)
{
    __shared__ float t[32][33];
    const int c0 = blockIdx.x * 32, r0 = blockIdx.y * 32;
    const int tx = threadIdx.x & 31, ty = threadIdx.x >> 5;
#pragma unroll
    for (int j = 0; j < 4; ++j)
        t[ty + 8 * j][tx] = in[(size_t)(r0 + ty + 8 * j) * C + c0 + tx];
    __syncthreads();
#pragma unroll
    for (int j = 0; j < 4; ++j)
        out[(size_t)(c0 + ty + 8 * j) * R + r0 + tx] = (__bf16)t[tx][ty + 8 * j];
}

// ---------------------------------------------------------------------------
// Causal row softmax in place (f32) + bf16 copy for the attn*V GEMM.
// ---------------------------------------------------------------------------
__global__ __launch_bounds__(256)
void softmax_causal_kernel(float* __restrict__ attn, __bf16* __restrict__ attb, int S)
{
    __shared__ float red[256];
    const int tid = threadIdx.x;
    const int r   = (int)(blockIdx.x % (unsigned)S);
    float*  p  = attn + (size_t)blockIdx.x * S;
    __bf16* pb = attb + (size_t)blockIdx.x * S;
    const int valid = r + 1;

    float mx = -3.402823466e38f;
    for (int c = tid; c < valid; c += 256) mx = fmaxf(mx, p[c]);
    red[tid] = mx; __syncthreads();
#pragma unroll
    for (int o = 128; o > 0; o >>= 1) {
        if (tid < o) red[tid] = fmaxf(red[tid], red[tid + o]);
        __syncthreads();
    }
    mx = red[0]; __syncthreads();

    float sm = 0.0f;
    for (int c = tid; c < valid; c += 256) sm += __expf(p[c] - mx);
    red[tid] = sm; __syncthreads();
#pragma unroll
    for (int o = 128; o > 0; o >>= 1) {
        if (tid < o) red[tid] += red[tid + o];
        __syncthreads();
    }
    const float inv = 1.0f / red[0];

    for (int c = tid; c < valid; c += 256) {
        const float e = __expf(p[c] - mx) * inv;
        p[c]  = e;
        pb[c] = (__bf16)e;
    }
    for (int c = valid + tid; c < S; c += 256) { p[c] = 0.0f; pb[c] = (__bf16)0.0f; }
}

// ---------------------------------------------------------------------------
// out = LN(A + Bv; g,b,eps) [+ resid]; optional bf16 copy of out.
// ---------------------------------------------------------------------------
__global__ __launch_bounds__(256)
void add_ln_kernel(const float* __restrict__ A, const float* __restrict__ Bv,
                   const float* __restrict__ g, const float* __restrict__ b,
                   float* __restrict__ outp, __bf16* __restrict__ out16,
                   const float* __restrict__ resid, int Dd, float eps)
{
    __shared__ float red[256];
    __shared__ float red2[256];
    const int tid = threadIdx.x;
    const size_t row = blockIdx.x;
    const float* pa = A  + row * Dd;
    const float* pb = Bv + row * Dd;

    float s = 0.0f, s2 = 0.0f;
    for (int c = tid; c < Dd; c += 256) {
        const float v = pa[c] + pb[c];
        s += v; s2 += v * v;
    }
    red[tid] = s; red2[tid] = s2; __syncthreads();
#pragma unroll
    for (int o = 128; o > 0; o >>= 1) {
        if (tid < o) { red[tid] += red[tid + o]; red2[tid] += red2[tid + o]; }
        __syncthreads();
    }
    const float mu   = red[0] / (float)Dd;
    const float var  = red2[0] / (float)Dd - mu * mu;
    const float rstd = rsqrtf(var + eps);

    const float* pr = resid ? (resid + row * Dd) : nullptr;
    for (int c = tid; c < Dd; c += 256) {
        const float v = pa[c] + pb[c];
        float y = (v - mu) * rstd * g[c] + b[c];
        if (pr) y += pr[c];
        outp[row * Dd + c] = y;
        if (out16) out16[row * Dd + c] = (__bf16)y;
    }
}

// ---------------------------------------------------------------------------
extern "C" void kernel_launch(void* const* d_in, const int* in_sizes, int n_in,
                              void* d_out, int out_size, void* d_ws, size_t ws_size,
                              hipStream_t stream)
{
    (void)in_sizes; (void)n_in; (void)out_size; (void)ws_size;

    const float* x    = (const float*)d_in[0];
    /* d_in[1] = causal mask: handled analytically (tril) */
    const float* w_q  = (const float*)d_in[2];
    const float* w_k  = (const float*)d_in[3];
    const float* w_v  = (const float*)d_in[4];
    const float* ln1g = (const float*)d_in[5];
    const float* ln1b = (const float*)d_in[6];
    const float* w1   = (const float*)d_in[7];
    const float* w2   = (const float*)d_in[8];
    const float* ln2g = (const float*)d_in[9];
    const float* ln2b = (const float*)d_in[10];

    float* out  = (float*)d_out;
    float* attn = out + (size_t)TB * TS * TD;

    const size_t MS = (size_t)TB * TS;              // 4096 rows
    const size_t eD = MS * TD;                      // 8.4M elems

    // ---- workspace layout (bf16 operand pool + f32 activations) ----------
    __bf16* xb   = (__bf16*)d_ws;                   // [MS][D]
    __bf16* Qb   = xb   + eD;                       // [MS][D]
    __bf16* Kb   = Qb   + eD;                       // [MS][D]   (K-major for B)
    __bf16* Vt   = Kb   + eD;                       // [D][MS]   (transposed V)
    __bf16* wqT  = Vt   + eD;                       // [D][D]
    __bf16* wkT  = wqT  + (size_t)TD * TD;
    __bf16* wvT  = wkT  + (size_t)TD * TD;
    __bf16* w1T  = wvT  + (size_t)TD * TD;          // [F][D]
    __bf16* w2T  = w1T  + (size_t)TD * TF;          // [D][F]
    __bf16* hb   = w2T  + (size_t)TD * TF;          // [MS][D]
    __bf16* actb = hb   + eD;                       // [MS][F]
    __bf16* attb = actb + MS * TF;                  // [B,H,S,S] bf16
    float* O     = (float*)(attb + (size_t)TB * THH * TS * TS);  // [MS][D]
    float* h     = O + eD;                          // [MS][D]
    float* ff    = h + eD;                          // [MS][D]

    const dim3 blk(256);
    const long long SD = (long long)TS * TD;
    const long long SS = (long long)TS * TS;
    const float qscale = 0.08838834764831845f;      // 1/sqrt(DH)

    // ---- one-time operand prep (bf16 / transposed bf16) -------------------
    cvt_bf16_kernel<<<(unsigned)(eD / 4 / 256), blk, 0, stream>>>(x, xb, (long long)(eD / 4));
    transpose_cvt_kernel<<<dim3(TD / 32, TD / 32), blk, 0, stream>>>(w_q, wqT, TD, TD);
    transpose_cvt_kernel<<<dim3(TD / 32, TD / 32), blk, 0, stream>>>(w_k, wkT, TD, TD);
    transpose_cvt_kernel<<<dim3(TD / 32, TD / 32), blk, 0, stream>>>(w_v, wvT, TD, TD);
    transpose_cvt_kernel<<<dim3(TF / 32, TD / 32), blk, 0, stream>>>(w1, w1T, TD, TF);
    transpose_cvt_kernel<<<dim3(TD / 32, TF / 32), blk, 0, stream>>>(w2, w2T, TF, TD);

    // ---- QKV projections (TDM-staged bf16), Q/K bf16, V transposed bf16 ---
    gemm_wmma<true, false, false, false, 1><<<dim3(TD / 128, MS / 128, 1), blk, 0, stream>>>(
        xb, TD, 0, 0, wqT, TD, 0, 0, Qb, TD, 0, 0, TD, 1, 1.0f);
    gemm_wmma<true, false, false, false, 1><<<dim3(TD / 128, MS / 128, 1), blk, 0, stream>>>(
        xb, TD, 0, 0, wkT, TD, 0, 0, Kb, TD, 0, 0, TD, 1, 1.0f);
    gemm_wmma<true, false, false, false, 2><<<dim3(TD / 128, MS / 128, 1), blk, 0, stream>>>(
        xb, TD, 0, 0, wvT, TD, 0, 0, Vt, (long long)MS, 0, 0, TD, 1, 1.0f);

    // ---- scores = scale * Q K^T per (b,h); upper-tri tiles skipped --------
    gemm_wmma<true, false, true, false, 0><<<dim3(TS / 128, TS / 128, TB * THH), blk, 0, stream>>>(
        Qb, TD, SD, TDH, Kb, TD, SD, TDH,
        attn, TS, (long long)THH * SS, SS, TDH, THH, qscale);

    // ---- causal softmax (f32 out + bf16 operand copy) ---------------------
    softmax_causal_kernel<<<TB * THH * TS, blk, 0, stream>>>(attn, attb, TS);

    // ---- O = attn @ V per (b,h); causal K-limit; B = Vt via TDM -----------
    gemm_wmma<true, false, false, true, 0><<<dim3(TDH / 128, TS / 128, TB * THH), blk, 0, stream>>>(
        attb, TS, (long long)THH * SS, SS, Vt, (long long)MS, TS, (long long)TDH * MS,
        O, TD, SD, TDH, TS, THH, 1.0f);

    // ---- h = LN1(O + x) (f32 + bf16 copy) ---------------------------------
    add_ln_kernel<<<(unsigned)MS, blk, 0, stream>>>(O, x, ln1g, ln1b, h, hb, nullptr, TD, 1e-5f);

    // ---- FFN: actb = relu(h @ w1) bf16;  ff = actb @ w2 f32 ---------------
    gemm_wmma<true, true, false, false, 1><<<dim3(TF / 128, MS / 128, 1), blk, 0, stream>>>(
        hb, TD, 0, 0, w1T, TD, 0, 0, actb, TF, 0, 0, TD, 1, 1.0f);
    gemm_wmma<true, false, false, false, 0><<<dim3(TD / 128, MS / 128, 1), blk, 0, stream>>>(
        actb, TF, 0, 0, w2T, TF, 0, 0, ff, TD, 0, 0, TF, 1, 1.0f);

    // ---- out = h + LN2(ff + h) --------------------------------------------
    add_ln_kernel<<<(unsigned)MS, blk, 0, stream>>>(ff, h, ln2g, ln2b, out, nullptr, h, TD, 1e-6f);
}
